// GraphHAM_50148038148194
// MI455X (gfx1250) — compile-verified, matrix operand order checked by reading, not run on hardware
//
#include <hip/hip_runtime.h>
#include <hip/hip_bf16.h>

// ---------------- problem constants ----------------
#define NN 10000
#define TT 160000
#define DD 256
#define PP 256
#define HH 512
#define SS 64

typedef __attribute__((ext_vector_type(16))) __bf16 v16bf;
typedef __attribute__((ext_vector_type(8)))  __bf16 v8bf;
typedef __attribute__((ext_vector_type(4)))  __bf16 v4bf;
typedef __attribute__((ext_vector_type(8)))  float  v8f;

// ---------------- async global->LDS staging (CDNA5 path, guarded) ----------------
#if defined(__AMDGCN__) && __has_builtin(__builtin_amdgcn_global_load_async_to_lds_b128) && __has_builtin(__builtin_amdgcn_s_wait_asynccnt)
#define ASYNC_LDS 1
typedef int v4i_t __attribute__((vector_size(16)));
typedef __attribute__((address_space(1))) v4i_t* gv4i_p;   // global int4*
typedef __attribute__((address_space(3))) v4i_t* lv4i_p;   // LDS int4*
__device__ __forceinline__ void async_cp16(const float* g, float* l) {
    __builtin_amdgcn_global_load_async_to_lds_b128(
        (gv4i_p)(unsigned long long)g,
        (lv4i_p)(unsigned int)(unsigned long long)l,
        /*offset=*/0, /*cpol=*/0);
}
__device__ __forceinline__ void async_wait_all() { __builtin_amdgcn_s_wait_asynccnt(0); }
#else
#define ASYNC_LDS 0
__device__ __forceinline__ void async_cp16(const float* g, float* l) {
    *(float4*)l = *(const float4*)g;
}
__device__ __forceinline__ void async_wait_all() {}
#endif

// ---------------- WMMA helpers ----------------
__device__ __forceinline__ v8f wmma_bf(v16bf a, v16bf b, v8f c) {
    // D = A(16x32 bf16) * B(32x16 bf16) + C(16x16 f32)
    return __builtin_amdgcn_wmma_f32_16x16x32_bf16(
        /*neg_a=*/false, a, /*neg_b=*/false, b,
        /*c_mod=*/(short)0, c, /*reuse_a=*/false, /*reuse_b=*/false);
}

// A-tile from LDS row-major bf16 [16 x ldx], K-tile kt (32 wide).
// ISA layout: lane<16 -> M=lane, element i -> K = (i>>3)*16 + (i&7) + (lane>=16? 8:0)
__device__ __forceinline__ v16bf load_a_lds(const __bf16* Xs, int ldx, int kt, int lane) {
    const __bf16* p = Xs + (lane & 15) * ldx + (kt << 5) + ((lane >> 4) << 3);
    v8bf lo = *(const v8bf*)(p);        // K offsets 0..7
    v8bf hi = *(const v8bf*)(p + 16);   // K offsets 16..23
    return __builtin_shufflevector(lo, hi, 0,1,2,3,4,5,6,7,8,9,10,11,12,13,14,15);
}

// A-tile from LDS row-major f32 [16 x ldx] with on-the-fly bf16 convert
__device__ __forceinline__ v16bf load_a_lds_f32(const float* Xs, int ldx, int kt, int lane) {
    const float* p = Xs + (lane & 15) * ldx + (kt << 5) + ((lane >> 4) << 3);
    float4 q0 = *(const float4*)(p);
    float4 q1 = *(const float4*)(p + 4);
    float4 q2 = *(const float4*)(p + 16);
    float4 q3 = *(const float4*)(p + 20);
    v16bf a;
    a[0]  = (__bf16)q0.x; a[1]  = (__bf16)q0.y; a[2]  = (__bf16)q0.z; a[3]  = (__bf16)q0.w;
    a[4]  = (__bf16)q1.x; a[5]  = (__bf16)q1.y; a[6]  = (__bf16)q1.z; a[7]  = (__bf16)q1.w;
    a[8]  = (__bf16)q2.x; a[9]  = (__bf16)q2.y; a[10] = (__bf16)q2.z; a[11] = (__bf16)q2.w;
    a[12] = (__bf16)q3.x; a[13] = (__bf16)q3.y; a[14] = (__bf16)q3.z; a[15] = (__bf16)q3.w;
    return a;
}

// B-tile pre-swizzled: contiguous 16 bf16 per (tile, lane)
__device__ __forceinline__ v16bf load_b_pre(const __bf16* B, int ktiles, int nt, int kt, int lane) {
    const __bf16* p = B + (((size_t)(nt * ktiles + kt) << 5) + lane) * 16;
    v8bf lo = *(const v8bf*)(p);
    v8bf hi = *(const v8bf*)(p + 8);
    return __builtin_shufflevector(lo, hi, 0,1,2,3,4,5,6,7,8,9,10,11,12,13,14,15);
}

__device__ __forceinline__ void zero_acc(v8f& a) {
#pragma unroll
    for (int i = 0; i < 8; ++i) a[i] = 0.0f;
}

// ---------------- weight pre-swizzle: f32 [K x Nw] -> bf16 B-operand layout ----------------
__global__ void k_preconv_b(const float* __restrict__ W, int K, int Nw, __bf16* __restrict__ out) {
    int idx = blockIdx.x * blockDim.x + threadIdx.x;   // one thread per (tile, lane)
    int ktiles = K >> 5;
    int total = (Nw >> 4) * ktiles * 32;
    if (idx >= total) return;
    int lane = idx & 31;
    int tile = idx >> 5;
    int kt = tile % ktiles;
    int nt = tile / ktiles;
    int n  = (nt << 4) + (lane & 15);
    int kb = (kt << 5) + ((lane >> 4) << 3);
    __bf16* o = out + (size_t)idx * 16;
#pragma unroll
    for (int i = 0; i < 16; ++i) {
        int k = kb + ((i >> 3) << 4) + (i & 7);
        o[i] = (__bf16)W[(size_t)k * Nw + n];
    }
}

__global__ void k_zero4(float4* __restrict__ p, int n4) {
    int i = blockIdx.x * blockDim.x + threadIdx.x;
    if (i < n4) p[i] = float4{0.f, 0.f, 0.f, 0.f};
}

// ---------------- center = feat_A @ type_weight  [N,256]x[256,256] ----------------
__global__ void __launch_bounds__(128) k_center(const float* __restrict__ X,
                                                const __bf16* __restrict__ Bw,
                                                float* __restrict__ Cout) {
    constexpr int LDXF = DD + 8;                 // 264 f32 -> 1056B row, 16B aligned
    __shared__ __align__(16) float xsh[16 * LDXF];
    const int row0 = blockIdx.x * 16;
    const int tid = threadIdx.x, lane = tid & 31, wave = tid >> 5;

    constexpr int CH = DD / 4;                   // float4 chunks per row
    for (int idx = tid; idx < 16 * CH; idx += 128) {
        int r = idx / CH, c = (idx - r * CH) << 2;
        async_cp16(X + (size_t)(row0 + r) * DD + c, &xsh[r * LDXF + c]);
    }
    async_wait_all();
    __syncthreads();

    v8f acc[4];
#pragma unroll
    for (int t = 0; t < 4; ++t) zero_acc(acc[t]);
    for (int kt = 0; kt < DD / 32; ++kt) {
        v16bf a = load_a_lds_f32(xsh, LDXF, kt, lane);
#pragma unroll
        for (int t = 0; t < 4; ++t) {
            v16bf b = load_b_pre(Bw, DD / 32, wave * 4 + t, kt, lane);
            acc[t] = wmma_bf(a, b, acc[t]);
        }
    }
    const int mbase = (lane >> 4) << 3;
#pragma unroll
    for (int t = 0; t < 4; ++t) {
        int n = (wave * 4 + t) * 16 + (lane & 15);
#pragma unroll
        for (int r = 0; r < 8; ++r)
            Cout[(size_t)(row0 + mbase + r) * PP + n] = acc[t][r];
    }
}

// ---------------- fused metapath MLP + segment-sum ----------------
// dec = relu(X @ We + be) @ Wd + bd ; inj[seg[t]] += dec[t]
template <int DIN>
__global__ void __launch_bounds__(128) k_mlp(const float* __restrict__ X,
                                             const __bf16* __restrict__ Benc,
                                             const float* __restrict__ be,
                                             const __bf16* __restrict__ Bdec,
                                             const float* __restrict__ bd,
                                             const int* __restrict__ seg,
                                             float* __restrict__ inj) {
    constexpr int LDXF = DIN + 8;
    constexpr int LDH = HH + 8;                  // 520
    __shared__ __align__(16) float  xsh[16 * LDXF];
    __shared__ __align__(16) __bf16 hsh[16 * LDH];
    const int row0 = blockIdx.x * 16;
    const int tid = threadIdx.x, lane = tid & 31, wave = tid >> 5;

    constexpr int CH = DIN / 4;
    for (int idx = tid; idx < 16 * CH; idx += 128) {
        int r = idx / CH, c = (idx - r * CH) << 2;
        async_cp16(X + (size_t)(row0 + r) * DIN + c, &xsh[r * LDXF + c]);
    }
    async_wait_all();
    __syncthreads();

    // ---- encode: H=512 -> 32 N-tiles, 8 per wave ----
    constexpr int KT1 = DIN / 32;
    v8f acc[8];
#pragma unroll
    for (int t = 0; t < 8; ++t) zero_acc(acc[t]);
    for (int kt = 0; kt < KT1; ++kt) {
        v16bf a = load_a_lds_f32(xsh, LDXF, kt, lane);
#pragma unroll
        for (int t = 0; t < 8; ++t) {
            v16bf b = load_b_pre(Benc, KT1, wave * 8 + t, kt, lane);
            acc[t] = wmma_bf(a, b, acc[t]);
        }
    }
    const int mbase = (lane >> 4) << 3;
#pragma unroll
    for (int t = 0; t < 8; ++t) {
        int n = (wave * 8 + t) * 16 + (lane & 15);
        float bias = be[n];
#pragma unroll
        for (int r = 0; r < 8; ++r) {
            float v = acc[t][r] + bias;
            hsh[(mbase + r) * LDH + n] = (__bf16)(v > 0.0f ? v : 0.0f);
        }
    }
    __syncthreads();

    // ---- decode: P=256 -> 16 N-tiles, 4 per wave; K=512 -> 16 k-tiles ----
    v8f acc2[4];
#pragma unroll
    for (int t = 0; t < 4; ++t) zero_acc(acc2[t]);
    for (int kt = 0; kt < HH / 32; ++kt) {
        v16bf a = load_a_lds(hsh, LDH, kt, lane);
#pragma unroll
        for (int t = 0; t < 4; ++t) {
            v16bf b = load_b_pre(Bdec, HH / 32, wave * 4 + t, kt, lane);
            acc2[t] = wmma_bf(a, b, acc2[t]);
        }
    }
    int segs[8];
#pragma unroll
    for (int r = 0; r < 8; ++r) segs[r] = seg[row0 + mbase + r];
#pragma unroll
    for (int t = 0; t < 4; ++t) {
        int n = (wave * 4 + t) * 16 + (lane & 15);
        float bias = bd[n];
#pragma unroll
        for (int r = 0; r < 8; ++r)
            atomicAdd(&inj[(size_t)segs[r] * PP + n], acc2[t][r] + bias);
    }
}

// ---------------- attention score: mean(leaky_relu((tanh([c,inj]) @ W1) @ W2)) ----------------
__global__ void __launch_bounds__(128) k_attn(const float* __restrict__ Cen,
                                              const float* __restrict__ Inj,
                                              const __bf16* __restrict__ B1,
                                              const float* __restrict__ W2,
                                              float* __restrict__ score_acc) {
    constexpr int K2 = 2 * PP;                   // 512
    constexpr int LDX = K2 + 8;                  // 520
    __shared__ __align__(16) __bf16 xsh[16 * LDX];
    __shared__ float ysh[16 * LDX];
    const int row0 = blockIdx.x * 16;
    const int tid = threadIdx.x, lane = tid & 31, wave = tid >> 5;

    // stage tanh(concat(center, inj)) as bf16; float4 loads, 4x bf16 stores
    for (int idx = tid; idx < 16 * (K2 / 4); idx += 128) {
        int r = idx >> 7, c = (idx & 127) << 2;
        const float* src = (c < PP) ? &Cen[(size_t)(row0 + r) * PP + c]
                                    : &Inj[(size_t)(row0 + r) * PP + (c - PP)];
        float4 v = *(const float4*)src;
        v4bf o;
        o[0] = (__bf16)tanhf(v.x); o[1] = (__bf16)tanhf(v.y);
        o[2] = (__bf16)tanhf(v.z); o[3] = (__bf16)tanhf(v.w);
        *(v4bf*)&xsh[r * LDX + c] = o;
    }
    __syncthreads();

    v8f acc[8];
#pragma unroll
    for (int t = 0; t < 8; ++t) zero_acc(acc[t]);
    for (int kt = 0; kt < K2 / 32; ++kt) {
        v16bf a = load_a_lds(xsh, LDX, kt, lane);
#pragma unroll
        for (int t = 0; t < 8; ++t) {
            v16bf b = load_b_pre(B1, K2 / 32, wave * 8 + t, kt, lane);
            acc[t] = wmma_bf(a, b, acc[t]);
        }
    }
    const int mbase = (lane >> 4) << 3;
#pragma unroll
    for (int t = 0; t < 8; ++t) {
        int n = (wave * 8 + t) * 16 + (lane & 15);
#pragma unroll
        for (int r = 0; r < 8; ++r) ysh[(mbase + r) * LDX + n] = acc[t][r];
    }
    __syncthreads();

    if (tid < 16) {
        float s = 0.0f;
        for (int n = 0; n < K2; ++n) s += ysh[tid * LDX + n] * W2[n];
        s = (s > 0.0f) ? s : 0.1f * s;           // LeakyReLU(0.1)
        atomicAdd(score_acc, s);
    }
}

// ---------------- softmax over [0, s1, s2] ----------------
__global__ void k_softmax3(const float* __restrict__ accs, float* __restrict__ wts) {
    float s1 = accs[0] / (float)NN;
    float s2 = accs[1] / (float)NN;
    float m  = fmaxf(0.0f, fmaxf(s1, s2));
    float e0 = expf(-m), e1 = expf(s1 - m), e2 = expf(s2 - m);
    float inv = 1.0f / (e0 + e1 + e2);
    wts[0] = e1 * inv;                           // weight for inj_AP
    wts[1] = e2 * inv;                           // weight for inj_APA
}

__global__ void k_scale4(const float4* __restrict__ inj, const float* __restrict__ wptr,
                         float4* __restrict__ out, int n4) {
    int i = blockIdx.x * blockDim.x + threadIdx.x;
    if (i < n4) {
        float w = wptr[0];
        float4 v = inj[i];
        out[i] = float4{v.x * w, v.y * w, v.z * w, v.w * w};
    }
}

// ---------------- final: log_softmax((center + embed) @ W_cls + b) ----------------
__global__ void __launch_bounds__(128) k_final(const float* __restrict__ Cen,
                                               const float* __restrict__ injAP,
                                               const float* __restrict__ injAPA,
                                               const float* __restrict__ wts,
                                               const __bf16* __restrict__ Bcls,
                                               const float* __restrict__ bcls,
                                               float* __restrict__ out) {
    constexpr int LDX = DD + 8;
    __shared__ __align__(16) __bf16 xsh[16 * LDX];
    __shared__ float zsh[16][SS + 1];
    const int row0 = blockIdx.x * 16;
    const int tid = threadIdx.x, lane = tid & 31, wave = tid >> 5;
    const float w1 = wts[0], w2 = wts[1];

    for (int idx = tid; idx < 16 * (DD / 4); idx += 128) {
        int r = idx >> 6, c = (idx & 63) << 2;
        size_t o = (size_t)(row0 + r) * PP + c;
        float4 a = *(const float4*)&Cen[o];
        float4 b = *(const float4*)&injAP[o];
        float4 d = *(const float4*)&injAPA[o];
        v4bf ov;
        ov[0] = (__bf16)(a.x + 0.5f * (w1 * b.x + w2 * d.x));
        ov[1] = (__bf16)(a.y + 0.5f * (w1 * b.y + w2 * d.y));
        ov[2] = (__bf16)(a.z + 0.5f * (w1 * b.z + w2 * d.z));
        ov[3] = (__bf16)(a.w + 0.5f * (w1 * b.w + w2 * d.w));
        *(v4bf*)&xsh[r * LDX + c] = ov;
    }
    __syncthreads();

    v8f acc;
    zero_acc(acc);
    for (int kt = 0; kt < DD / 32; ++kt) {
        v16bf a = load_a_lds(xsh, LDX, kt, lane);
        v16bf b = load_b_pre(Bcls, DD / 32, wave, kt, lane);
        acc = wmma_bf(a, b, acc);
    }
    const int mbase = (lane >> 4) << 3;
    {
        int n = wave * 16 + (lane & 15);
        float bias = bcls[n];
#pragma unroll
        for (int r = 0; r < 8; ++r) zsh[mbase + r][n] = acc[r] + bias;
    }
    __syncthreads();

    if (tid < 16) {
        float mx = zsh[tid][0];
        for (int j = 1; j < SS; ++j) mx = fmaxf(mx, zsh[tid][j]);
        float sum = 0.0f;
        for (int j = 0; j < SS; ++j) sum += expf(zsh[tid][j] - mx);
        float lse = mx + logf(sum);
        for (int j = 0; j < SS; ++j)
            out[(size_t)(row0 + tid) * SS + j] = zsh[tid][j] - lse;
    }
}

// ---------------- host orchestration ----------------
extern "C" void kernel_launch(void* const* d_in, const int* in_sizes, int n_in,
                              void* d_out, int out_size, void* d_ws, size_t ws_size,
                              hipStream_t stream) {
    const float* feat_A    = (const float*)d_in[0];
    const float* feat_AP   = (const float*)d_in[1];
    const float* feat_APA  = (const float*)d_in[2];
    const int*   seg_AP    = (const int*)d_in[3];
    const int*   seg_APA   = (const int*)d_in[4];
    const float* type_w    = (const float*)d_in[5];
    const float* W_enc_AP  = (const float*)d_in[6];
    const float* b_enc_AP  = (const float*)d_in[7];
    const float* W_dec_AP  = (const float*)d_in[8];
    const float* b_dec_AP  = (const float*)d_in[9];
    const float* W_enc_APA = (const float*)d_in[10];
    const float* b_enc_APA = (const float*)d_in[11];
    const float* W_dec_APA = (const float*)d_in[12];
    const float* b_dec_APA = (const float*)d_in[13];
    const float* W_attn1   = (const float*)d_in[14];
    const float* W_attn2   = (const float*)d_in[15];
    const float* W_cls     = (const float*)d_in[16];
    const float* b_cls     = (const float*)d_in[17];

    // workspace layout (floats, then bf16 pre-swizzled weights)
    float* ws_f   = (float*)d_ws;
    float* center = ws_f;                               // N*P
    float* injAP  = center + (size_t)NN * PP;           // N*P
    float* injAPA = injAP + (size_t)NN * PP;            // N*P
    float* scal   = injAPA + (size_t)NN * PP;           // [0]=sAP [1]=sAPA [2]=wAP [3]=wAPA (+pad to 64)
    __bf16* Btw     = (__bf16*)(scal + 64);             // 32B-aligned byte offset
    __bf16* BencAP  = Btw     + 256 * 256;
    __bf16* BdecAP  = BencAP  + 256 * 512;
    __bf16* BencAPA = BdecAP  + 512 * 256;
    __bf16* BdecAPA = BencAPA + 512 * 512;
    __bf16* Battn1  = BdecAPA + 512 * 256;
    __bf16* Bcls    = Battn1  + 512 * 512;

    auto pc = [&](const float* W, int K, int Nw, __bf16* out) {
        int threads = (Nw / 16) * (K / 32) * 32;
        k_preconv_b<<<(threads + 255) / 256, 256, 0, stream>>>(W, K, Nw, out);
    };
    pc(type_w,    256, 256, Btw);
    pc(W_enc_AP,  256, 512, BencAP);
    pc(W_dec_AP,  512, 256, BdecAP);
    pc(W_enc_APA, 512, 512, BencAPA);
    pc(W_dec_APA, 512, 256, BdecAPA);
    pc(W_attn1,   512, 512, Battn1);
    pc(W_cls,     256,  64, Bcls);

    int nz4 = (2 * NN * PP + 64) / 4;                   // inj buffers + scalars
    k_zero4<<<(nz4 + 255) / 256, 256, 0, stream>>>((float4*)injAP, nz4);

    k_center<<<NN / 16, 128, 0, stream>>>(feat_A, Btw, center);

    k_mlp<256><<<TT / 16, 128, 0, stream>>>(feat_AP,  BencAP,  b_enc_AP,  BdecAP,  b_dec_AP,  seg_AP,  injAP);
    k_mlp<512><<<TT / 16, 128, 0, stream>>>(feat_APA, BencAPA, b_enc_APA, BdecAPA, b_dec_APA, seg_APA, injAPA);

    k_attn<<<NN / 16, 128, 0, stream>>>(center, injAP,  Battn1, W_attn2, &scal[0]);
    k_attn<<<NN / 16, 128, 0, stream>>>(center, injAPA, Battn1, W_attn2, &scal[1]);

    k_softmax3<<<1, 1, 0, stream>>>(scal, scal + 2);

    float* outF = (float*)d_out;
    int np4 = NN * PP / 4;
    k_scale4<<<(np4 + 255) / 256, 256, 0, stream>>>((const float4*)injAP,  scal + 2,
                                                    (float4*)(outF + (size_t)NN * SS), np4);
    k_scale4<<<(np4 + 255) / 256, 256, 0, stream>>>((const float4*)injAPA, scal + 3,
                                                    (float4*)(outF + (size_t)NN * SS + (size_t)NN * PP), np4);

    k_final<<<NN / 16, 128, 0, stream>>>(center, injAP, injAPA, scal + 2, Bcls, b_cls, outF);
}
